// get_model_49778670961215
// MI455X (gfx1250) — compile-verified, matrix operand order checked by reading, not run on hardware
//
#include <hip/hip_runtime.h>
#include <cstdint>

// ---------------------------------------------------------------------------
// PCT SA_Layer (offset attention) for MI455X / gfx1250, wave32 + WMMA.
// B=16, C=128, N=2048. All GEMMs use v_wmma_f32_16x16x32_f16.
// Attention matrix is never materialized in HBM (saves ~512MB of traffic ->
// ~22us at 23.3 TB/s, more than the total compute time). Energy (K=32, one
// WMMA per 16x16 tile) is recomputed in the x_r pass instead.
// ---------------------------------------------------------------------------

#define Bv 16
#define Cv 128
#define Nv 2048
#define Dv 32 // C/4

typedef __attribute__((ext_vector_type(16))) _Float16 v16h;
typedef __attribute__((ext_vector_type(8)))  float    v8f;

// CDNA5 16-bit A/B operand K-pattern (ISA 7.12.2):
// lane<16 : halves 0..7 -> K=0..7,  halves 8..15 -> K=16..23
// lane>=16: halves 0..7 -> K=8..15, halves 8..15 -> K=24..31
__device__ __forceinline__ int kpat(int h, int hi) {
  return ((h < 8) ? h : (h + 8)) + (hi ? 8 : 0);
}

// Operand tile, element (k, j) at p[k*ldk + j]; j = M (for A) or N (for B).
// Across lanes (fixed k) these loads are contiguous.
__device__ __forceinline__ v16h ld_kmaj_h(const _Float16* __restrict__ p,
                                          int ldk, int j, int hi) {
  v16h r;
#pragma unroll
  for (int h = 0; h < 16; ++h) r[h] = p[(size_t)kpat(h, hi) * ldk + j];
  return r;
}
__device__ __forceinline__ v16h ld_kmaj_f(const float* __restrict__ p,
                                          int ldk, int j, int hi) {
  v16h r;
#pragma unroll
  for (int h = 0; h < 16; ++h) r[h] = (_Float16)p[(size_t)kpat(h, hi) * ldk + j];
  return r;
}
// Row-major weight tile, element (j, k) at p[j*ld + k] (tiny, cache-resident).
__device__ __forceinline__ v16h ld_rmaj_f(const float* __restrict__ p,
                                          int ld, int j, int hi) {
  v16h r;
#pragma unroll
  for (int h = 0; h < 16; ++h) r[h] = (_Float16)p[j * ld + kpat(h, hi)];
  return r;
}

__device__ __forceinline__ v8f wmma16(v16h a, v16h b, v8f c) {
  return __builtin_amdgcn_wmma_f32_16x16x32_f16(false, a, false, b,
                                                (short)0, c, false, false);
}

// ---------------------------------------------------------------------------
// K1: qkT[b][32][2048] = wq @ x (f16)   and   vT[b][2048][128] = (wv@x+bv)^T
// 1280 wave-tiles per batch, one 16x16 tile per wave, K=128 (4 WMMA steps).
// ---------------------------------------------------------------------------
__global__ __launch_bounds__(256) void k1_project(
    const float* __restrict__ x, const float* __restrict__ wq,
    const float* __restrict__ wv, const float* __restrict__ bvp,
    _Float16* __restrict__ qkT, _Float16* __restrict__ vT) {
  const int lane = threadIdx.x & 31, j = lane & 15, hi = lane >> 4;
  const int wid = (int)((blockIdx.x * blockDim.x + threadIdx.x) >> 5);
  const int b = wid / 1280;
  int t = wid % 1280;
  const float* xb = x + (size_t)b * Cv * Nv;
  if (t < 256) { // qkT tile: M=d(32 rows -> 2 tiles), N=n(128 tiles), K=c
    const int dt = t >> 7, nt = t & 127;
    v8f acc = {};
#pragma unroll
    for (int kk = 0; kk < 4; ++kk) {
      v16h a  = ld_rmaj_f(wq + (dt * 16) * Cv + kk * 32, Cv, j, hi);
      v16h bm = ld_kmaj_f(xb + (size_t)(kk * 32) * Nv + nt * 16, Nv, j, hi);
      acc = wmma16(a, bm, acc);
    }
    _Float16* q = qkT + (size_t)b * Dv * Nv;
#pragma unroll
    for (int v = 0; v < 8; ++v) // D: row = v+8*hi, col = j
      q[(size_t)(dt * 16 + v + 8 * hi) * Nv + nt * 16 + j] = (_Float16)acc[v];
  } else {       // vT tile: M=n, N=d, K=c (stores v transposed for K3's A-side)
    t -= 256;
    const int nt = t >> 3, dt = t & 7;
    const float bias = bvp[dt * 16 + j];
    v8f acc = {bias, bias, bias, bias, bias, bias, bias, bias};
#pragma unroll
    for (int kk = 0; kk < 4; ++kk) {
      v16h a  = ld_kmaj_f(xb + (size_t)(kk * 32) * Nv + nt * 16, Nv, j, hi);
      v16h bm = ld_rmaj_f(wv + (dt * 16) * Cv + kk * 32, Cv, j, hi);
      acc = wmma16(a, bm, acc);
    }
    _Float16* vb = vT + (size_t)b * Nv * Cv;
#pragma unroll
    for (int v = 0; v < 8; ++v)
      vb[(size_t)(nt * 16 + v + 8 * hi) * Cv + dt * 16 + j] = (_Float16)acc[v];
  }
}

// ---------------------------------------------------------------------------
// K2: per (b, 16-row block): energy row-block [16][2048] via WMMA into 128KB
// LDS (CDNA5 WGP has 320KB -> whole row-block fits; 2 blocks/WGP resident),
// then rowmax, rowsum, and atomic colsum partials. grid = B*128.
// ---------------------------------------------------------------------------
__global__ __launch_bounds__(256) void k2_softmax_stats(
    const _Float16* __restrict__ qkT, float* __restrict__ rowmax,
    float* __restrict__ rowsum, float* __restrict__ colsum) {
  extern __shared__ float e[]; // [16][Nv] f32 = 128 KB
  __shared__ float red[16][16];
  __shared__ float s_rmax[16], s_rsum[16];
  const int b = blockIdx.x >> 7, nb = blockIdx.x & 127;
  const int tid = threadIdx.x;
  const int lane = tid & 31, j = lane & 15, hi = lane >> 4, w = tid >> 5;
  const _Float16* qb = qkT + (size_t)b * Dv * Nv;

  // phase 1: energy tiles. A (q rows nb*16..+15) hoisted; K=32 -> 1 WMMA/tile.
  const v16h a = ld_kmaj_h(qb + nb * 16, Nv, j, hi);
  const v8f z = {};
  for (int i = 0; i < 16; ++i) {
    const int mt = w * 16 + i;
    v16h bm = ld_kmaj_h(qb + mt * 16, Nv, j, hi);
    v8f d = wmma16(a, bm, z);
#pragma unroll
    for (int v = 0; v < 8; ++v)
      e[(size_t)(v + 8 * hi) * Nv + mt * 16 + j] = d[v];
  }
  __syncthreads();

  // phase 2: rowmax / rowsum (16 threads per row, 128 cols each)
  const int r = tid >> 4, q = tid & 15;
  const float* er = e + (size_t)r * Nv;
  float lmax = -3.0e38f;
  for (int m = q * 128; m < q * 128 + 128; ++m) lmax = fmaxf(lmax, er[m]);
  red[r][q] = lmax;
  __syncthreads();
  if (q == 0) {
    float mx = red[r][0];
    for (int i = 1; i < 16; ++i) mx = fmaxf(mx, red[r][i]);
    s_rmax[r] = mx;
  }
  __syncthreads();
  const float rm = s_rmax[r];
  float ls = 0.f;
  for (int m = q * 128; m < q * 128 + 128; ++m) ls += __expf(er[m] - rm);
  red[r][q] = ls;
  __syncthreads();
  if (q == 0) {
    float s = 0.f;
    for (int i = 0; i < 16; ++i) s += red[r][i];
    s_rsum[r] = s;
    rowmax[b * Nv + nb * 16 + r] = rm;
    rowsum[b * Nv + nb * 16 + r] = s;
  }
  __syncthreads();

  // phase 3: column-sum partials of attn = exp(e-rmax)/rsum
  for (int m = tid; m < Nv; m += 256) {
    float s = 0.f;
#pragma unroll
    for (int rr = 0; rr < 16; ++rr)
      s += __expf(e[(size_t)rr * Nv + m] - s_rmax[rr]) / s_rsum[rr];
    atomicAdd(colsum + b * Nv + m, s);
  }
}

// ---------------------------------------------------------------------------
// K3: y = x - x_r, where x_r = v @ attn'. Flash-style: recompute energy
// (K=32 WMMA), build p = exp(e-rmax)/rsum in-register (energy-D layout maps
// 1:1 onto B-operand K-pattern per lane -> no cross-lane shuffles), then
// 16x16x32 WMMA against vT, finally scale by 1/(eps+colsum[m]).
// Block tile: 64 d x 128 m; wave: 64d x 16m. grid = B*2*16.
// ---------------------------------------------------------------------------
__global__ __launch_bounds__(256) void k3_xr(
    const float* __restrict__ x, const _Float16* __restrict__ qkT,
    const _Float16* __restrict__ vT, const float* __restrict__ rowmax,
    const float* __restrict__ rowsum, const float* __restrict__ colsum,
    _Float16* __restrict__ y) {
  __shared__ float s_rmax[Nv], s_rsum[Nv]; // 16 KB
  const int b = blockIdx.x >> 5;
  const int dblk = (blockIdx.x >> 4) & 1;
  const int mblk = blockIdx.x & 15;
  const int tid = threadIdx.x;
  const int lane = tid & 31, j = lane & 15, hi = lane >> 4, w = tid >> 5;
  for (int i = tid; i < Nv; i += 256) {
    s_rmax[i] = rowmax[b * Nv + i];
    s_rsum[i] = rowsum[b * Nv + i];
  }
  __syncthreads();
  const _Float16* qb = qkT + (size_t)b * Dv * Nv;
  const _Float16* vb = vT + (size_t)b * Nv * Cv;
  const int m0 = mblk * 128 + w * 16;
  const int d0 = dblk * 64;
  const v16h Bk = ld_kmaj_h(qb + m0, Nv, j, hi); // k columns m0..m0+15, hoisted
  const v8f z = {};
  v8f acc[4] = {z, z, z, z};
  for (int n0 = 0; n0 < Nv; n0 += 32) {
    __builtin_prefetch(qb + (n0 + 32), 0, 0); // global_prefetch_b8
    v16h Aq0 = ld_kmaj_h(qb + n0, Nv, j, hi);
    v16h Aq1 = ld_kmaj_h(qb + n0 + 16, Nv, j, hi);
    v8f E0 = wmma16(Aq0, Bk, z); // energy rows n0..n0+15
    v8f E1 = wmma16(Aq1, Bk, z); // energy rows n0+16..n0+31
    v16h Bp; // p tile as B operand: element (k=n_local, col j)
#pragma unroll
    for (int h = 0; h < 16; ++h) {
      const int rl = (h < 8) ? (h + 8 * hi) : (16 + (h - 8) + 8 * hi);
      const float ev = (h < 8) ? E0[h] : E1[h - 8]; // same lane, no shuffle
      const int n = n0 + rl;
      Bp[h] = (_Float16)(__expf(ev - s_rmax[n]) / s_rsum[n]);
    }
#pragma unroll
    for (int ds = 0; ds < 4; ++ds) {
      v16h Av = ld_kmaj_h(vb + (size_t)n0 * Cv + d0 + ds * 16, Cv, j, hi);
      acc[ds] = wmma16(Av, Bp, acc[ds]);
    }
  }
  const float inv = 1.0f / (1e-9f + colsum[b * Nv + m0 + j]);
  const float* xb = x + (size_t)b * Cv * Nv;
  _Float16* yb = y + (size_t)b * Cv * Nv;
#pragma unroll
  for (int ds = 0; ds < 4; ++ds)
#pragma unroll
    for (int v = 0; v < 8; ++v) {
      const int d = d0 + ds * 16 + v + 8 * hi;
      const int m = m0 + j;
      yb[(size_t)d * Nv + m] =
          (_Float16)(xb[(size_t)d * Nv + m] - acc[ds][v] * inv);
    }
}

// ---------------------------------------------------------------------------
// K4: out = x + relu(BN(wt @ y + bt)). Wave owns (ct, 4 n-tiles); weight
// A-tiles hoisted across the 4 n-tiles. grid = B*8*32 jobs / 8 waves.
// ---------------------------------------------------------------------------
__global__ __launch_bounds__(256) void k4_out(
    const float* __restrict__ x, const float* __restrict__ wt,
    const float* __restrict__ btp, const float* __restrict__ g_w,
    const float* __restrict__ b_w, const float* __restrict__ bn_mean,
    const float* __restrict__ bn_var, const _Float16* __restrict__ y,
    float* __restrict__ out) {
  const int lane = threadIdx.x & 31, j = lane & 15, hi = lane >> 4;
  const int wid = (int)((blockIdx.x * blockDim.x + threadIdx.x) >> 5);
  const int b = wid >> 8, ct = (wid >> 5) & 7, ng = wid & 31;
  const _Float16* yb = y + (size_t)b * Cv * Nv;
  const float* xb = x + (size_t)b * Cv * Nv;
  float* ob = out + (size_t)b * Cv * Nv;
  v16h Aw[4];
#pragma unroll
  for (int kk = 0; kk < 4; ++kk)
    Aw[kk] = ld_rmaj_f(wt + (ct * 16) * Cv + kk * 32, Cv, j, hi);
  float sc[8], sh[8], bias[8];
#pragma unroll
  for (int v = 0; v < 8; ++v) {
    const int c = ct * 16 + v + 8 * hi;
    const float s = g_w[c] * rsqrtf(bn_var[c] + 1e-5f);
    sc[v] = s;
    sh[v] = b_w[c] - bn_mean[c] * s;
    bias[v] = btp[c];
  }
  for (int st = 0; st < 4; ++st) {
    const int nt = ng * 4 + st;
    v8f acc;
#pragma unroll
    for (int v = 0; v < 8; ++v) acc[v] = bias[v];
#pragma unroll
    for (int kk = 0; kk < 4; ++kk) {
      v16h bm = ld_kmaj_h(yb + (size_t)(kk * 32) * Nv + nt * 16, Nv, j, hi);
      acc = wmma16(Aw[kk], bm, acc);
    }
#pragma unroll
    for (int v = 0; v < 8; ++v) {
      const int c = ct * 16 + v + 8 * hi;
      const int n = nt * 16 + j;
      const float t = acc[v] * sc[v] + sh[v];
      ob[(size_t)c * Nv + n] = xb[(size_t)c * Nv + n] + fmaxf(t, 0.f);
    }
  }
}

// ---------------------------------------------------------------------------
extern "C" void kernel_launch(void* const* d_in, const int* in_sizes, int n_in,
                              void* d_out, int out_size, void* d_ws,
                              size_t ws_size, hipStream_t stream) {
  (void)in_sizes; (void)n_in; (void)out_size; (void)ws_size;
  const float* x       = (const float*)d_in[0];
  const float* wq      = (const float*)d_in[1];
  const float* wv      = (const float*)d_in[2];
  const float* bvp     = (const float*)d_in[3];
  const float* wt      = (const float*)d_in[4];
  const float* btp     = (const float*)d_in[5];
  const float* g_w     = (const float*)d_in[6];
  const float* b_w     = (const float*)d_in[7];
  const float* bn_mean = (const float*)d_in[8];
  const float* bn_var  = (const float*)d_in[9];
  float* out = (float*)d_out;

  char* ws = (char*)d_ws;
  size_t off = 0;
  _Float16* qkT = (_Float16*)(ws + off); off += (size_t)Bv * Dv * Nv * 2;
  _Float16* vT  = (_Float16*)(ws + off); off += (size_t)Bv * Nv * Cv * 2;
  _Float16* y   = (_Float16*)(ws + off); off += (size_t)Bv * Cv * Nv * 2;
  float* rowmax = (float*)(ws + off);    off += (size_t)Bv * Nv * 4;
  float* rowsum = (float*)(ws + off);    off += (size_t)Bv * Nv * 4;
  float* colsum = (float*)(ws + off);    off += (size_t)Bv * Nv * 4;

  // colsum is accumulated atomically -> zero it every call (capture-safe).
  hipMemsetAsync(colsum, 0, (size_t)Bv * Nv * sizeof(float), stream);

  k1_project<<<2560, 256, 0, stream>>>(x, wq, wv, bvp, qkT, vT);

  // 128 KB dynamic LDS per workgroup (CDNA5 WGP has 320 KB).
  hipFuncSetAttribute(reinterpret_cast<const void*>(k2_softmax_stats),
                      hipFuncAttributeMaxDynamicSharedMemorySize, 131072);
  k2_softmax_stats<<<Bv * 128, 256, 131072, stream>>>(qkT, rowmax, rowsum,
                                                      colsum);
  k3_xr<<<Bv * 32, 256, 0, stream>>>(x, qkT, vT, rowmax, rowsum, colsum, y);
  k4_out<<<512, 256, 0, stream>>>(x, wt, btp, g_w, b_w, bn_mean, bn_var, y,
                                  out);
}
